// AttnBlock_27333171872001
// MI455X (gfx1250) — compile-verified, matrix operand order checked by reading, not run on hardware
//
#include <hip/hip_runtime.h>

typedef __bf16 bf16;
typedef __attribute__((ext_vector_type(16))) __bf16 v16bf;
typedef __attribute__((ext_vector_type(8)))  __bf16 v8bf;
typedef __attribute__((ext_vector_type(8)))  float  v8f;
typedef __attribute__((ext_vector_type(4)))  int    v4i;
typedef __attribute__((address_space(3))) v4i* lds_v4i_p;

#define Bq 8
#define Cq 512
#define Nq 4096
#define Mq 32768
#define SCALEF 0.044194173824159216f   // 1/sqrt(512)
#define EPSQ 1e-5f

// ---------------------------------------------------------------------------
// WMMA helpers
// ---------------------------------------------------------------------------
__device__ __forceinline__ v8f wmma_bf16(v16bf a, v16bf b, v8f c) {
  return __builtin_amdgcn_wmma_f32_16x16x32_bf16(false, a, false, b, (short)0, c,
                                                 false, false);
}

__device__ __forceinline__ v16bf cat8(v8bf lo, v8bf hi) {
  return __builtin_shufflevector(lo, hi, 0, 1, 2, 3, 4, 5, 6, 7,
                                         8, 9, 10, 11, 12, 13, 14, 15);
}

// Load a 16x32 16-bit A-style fragment from a row-major [ld] bf16 matrix.
// base points at (row0, k0). Per ISA layout: lane L -> row = L&15, half = L>>4;
// elements 0..7  -> k = half*8 + j        (contiguous 16B)
// elements 8..15 -> k = 16 + half*8 + j   (contiguous 16B)
// The same pattern serves as a B fragment when the source is the transpose
// (rows indexed by N, k contiguous along the row).
__device__ __forceinline__ v16bf frag_rowmajor(const bf16* base, int ld, int lane) {
  const bf16* p = base + (size_t)(lane & 15) * ld + ((lane >> 4) << 3);
  v8bf lo = *(const v8bf*)p;
  v8bf hi = *(const v8bf*)(p + 16);
  return cat8(lo, hi);
}

// ---------------------------------------------------------------------------
// Async global -> LDS staging (gfx1250 GLOBAL_LOAD_ASYNC_TO_LDS_B128,
// tracked by ASYNCcnt). Param 0 is a generic v4i* (per compiler diagnostic);
// LDS side is cast to addrspace(3), which also converts implicitly to
// generic if the builtin is declared that way.
// ---------------------------------------------------------------------------
__device__ __forceinline__ void stage16(const void* gsrc, void* ldst) {
#if __has_builtin(__builtin_amdgcn_global_load_async_to_lds_b128)
  __builtin_amdgcn_global_load_async_to_lds_b128((v4i*)gsrc, (lds_v4i_p)ldst, 0, 0);
#else
  *(uint4*)ldst = *(const uint4*)gsrc;
#endif
}

__device__ __forceinline__ void wait_async() {
#if __has_builtin(__builtin_amdgcn_s_wait_asynccnt)
  __builtin_amdgcn_s_wait_asynccnt(0);
#else
  asm volatile("s_wait_asynccnt 0" ::: "memory");
#endif
}

// ---------------------------------------------------------------------------
// 1) GroupNorm statistics: one block per (batch, group); 256-thread reduce.
// ---------------------------------------------------------------------------
__global__ __launch_bounds__(256) void k_gn_stats(const float* __restrict__ x,
                                                  float* __restrict__ stats) {
  int bg = blockIdx.x;            // 0..255 : b*32+g
  int b = bg >> 5, g = bg & 31;
  const float* xb = x + (size_t)b * Nq * Cq + g * 16;
  float s = 0.f, ss = 0.f;
  for (int n = threadIdx.x; n < Nq; n += 256) {
    const float* p = xb + (size_t)n * Cq;
#pragma unroll
    for (int ci = 0; ci < 16; ++ci) { float v = p[ci]; s += v; ss += v * v; }
  }
  __shared__ float sh_s[256], sh_ss[256];
  sh_s[threadIdx.x] = s; sh_ss[threadIdx.x] = ss;
  __syncthreads();
  for (int off = 128; off > 0; off >>= 1) {
    if (threadIdx.x < off) {
      sh_s[threadIdx.x]  += sh_s[threadIdx.x + off];
      sh_ss[threadIdx.x] += sh_ss[threadIdx.x + off];
    }
    __syncthreads();
  }
  if (threadIdx.x == 0) {
    float inv = 1.0f / (float)(Nq * 16);
    float mean = sh_s[0] * inv;
    float var  = sh_ss[0] * inv - mean * mean;
    stats[bg * 2]     = mean;
    stats[bg * 2 + 1] = rsqrtf(var + EPSQ);
  }
}

// ---------------------------------------------------------------------------
// 2) Apply GroupNorm, emit bf16 activations (row-major M x C).
// ---------------------------------------------------------------------------
__global__ __launch_bounds__(256) void k_gn_apply(const float* __restrict__ x,
                                                  const float* __restrict__ stats,
                                                  const float* __restrict__ gamma,
                                                  const float* __restrict__ beta,
                                                  bf16* __restrict__ hn) {
  size_t i4 = ((size_t)blockIdx.x * 256 + threadIdx.x) * 4;
  int c = (int)(i4 & (Cq - 1));
  int m = (int)(i4 >> 9);
  int b = m >> 12;
  int g = c >> 4;
  float mean = stats[(b * 32 + g) * 2];
  float rstd = stats[(b * 32 + g) * 2 + 1];
  float4 xv = *(const float4*)(x + i4);
  bf16* o = hn + i4;
  o[0] = (bf16)((xv.x - mean) * rstd * gamma[c]     + beta[c]);
  o[1] = (bf16)((xv.y - mean) * rstd * gamma[c + 1] + beta[c + 1]);
  o[2] = (bf16)((xv.z - mean) * rstd * gamma[c + 2] + beta[c + 2]);
  o[3] = (bf16)((xv.w - mean) * rstd * gamma[c + 3] + beta[c + 3]);
}

// ---------------------------------------------------------------------------
// 3) Pack a 512x512 f32 weight (w[k][n]) into WMMA-B-fragment layout:
//    wpk[(kt*32+nt)*32 + lane][16] bf16 -> one contiguous 32B load per frag.
// ---------------------------------------------------------------------------
__global__ __launch_bounds__(256) void k_pack_w(const float* __restrict__ w,
                                                bf16* __restrict__ wpk) {
  int tid = blockIdx.x * 256 + threadIdx.x;
  if (tid >= 16 * 32 * 32) return;   // ktiles(16) * ntiles(32) * lanes(32)
  int lane = tid & 31;
  int nt = (tid >> 5) & 31;
  int kt = tid >> 10;
  int n = nt * 16 + (lane & 15);
  int half = lane >> 4;
  bf16* dst = wpk + (size_t)tid * 16; // tid == (kt*32+nt)*32+lane
#pragma unroll
  for (int j = 0; j < 8; ++j)
    dst[j] = (bf16)w[(kt * 32 + half * 8 + j) * Cq + n];
#pragma unroll
  for (int j = 0; j < 8; ++j)
    dst[8 + j] = (bf16)w[(kt * 32 + 16 + half * 8 + j) * Cq + n];
}

// ---------------------------------------------------------------------------
// 4) Fused QKV projection. 8 waves share one async-staged 16x512 A tile;
//    each wave computes one 16x16 tile for q, k, v (A fragment reused 3x).
//    q,k stored row-major bf16; v stored transposed (C x N) for P*V frags.
// ---------------------------------------------------------------------------
__global__ __launch_bounds__(256) void k_qkv(const bf16* __restrict__ hn,
                                             const bf16* __restrict__ wq,
                                             const bf16* __restrict__ wk,
                                             const bf16* __restrict__ wv,
                                             const float* __restrict__ bq,
                                             const float* __restrict__ bk,
                                             const float* __restrict__ bv,
                                             bf16* __restrict__ qb,
                                             bf16* __restrict__ kb,
                                             bf16* __restrict__ vtb) {
  __shared__ __align__(16) bf16 atile[16 * 512];
  int wave = threadIdx.x >> 5, lane = threadIdx.x & 31;
  int mt = blockIdx.x >> 2;
  int ct = ((blockIdx.x & 3) << 3) + wave;
  int m0 = mt << 4;
  for (int i = threadIdx.x; i < 16 * 512 / 8; i += 256)
    stage16((const uint4*)(hn + (size_t)m0 * Cq) + i, (uint4*)atile + i);
  wait_async();
  __syncthreads();
  v8f aq = {}, ak = {}, av = {};
  for (int kt = 0; kt < 16; ++kt) {
    v16bf a = frag_rowmajor(atile + kt * 32, 512, lane);
    const v16bf* bpq = (const v16bf*)(wq + (size_t)((kt * 32 + ct) * 32 + lane) * 16);
    const v16bf* bpk = (const v16bf*)(wk + (size_t)((kt * 32 + ct) * 32 + lane) * 16);
    const v16bf* bpv = (const v16bf*)(wv + (size_t)((kt * 32 + ct) * 32 + lane) * 16);
    aq = wmma_bf16(a, *bpq, aq);
    ak = wmma_bf16(a, *bpk, ak);
    av = wmma_bf16(a, *bpv, av);
  }
  int half = lane >> 4;
  int col = (ct << 4) + (lane & 15);
  float xbq = bq[col], xbk = bk[col], xbv = bv[col];
#pragma unroll
  for (int r = 0; r < 8; ++r) {
    int row = m0 + r + half * 8;   // C/D layout: M = vgpr + half*8
    qb[(size_t)row * Cq + col] = (bf16)(aq[r] + xbq);
    kb[(size_t)row * Cq + col] = (bf16)(ak[r] + xbk);
    int bb = row >> 12, n = row & (Nq - 1);
    vtb[(size_t)bb * Cq * Nq + (size_t)col * Nq + n] = (bf16)(av[r] + xbv);
  }
}

// ---------------------------------------------------------------------------
// 5) Softmax statistics (pass A): one wave per 16-query tile.
//    WMMA score tiles + shfl_xor reductions across the 16-lane halves
//    (C/D layout: lane = key column, VGPR r = query row r + half*8).
// ---------------------------------------------------------------------------
__global__ __launch_bounds__(32) void k_attn_stats(const bf16* __restrict__ qb,
                                                   const bf16* __restrict__ kb,
                                                   float* __restrict__ ms) {
  __shared__ __align__(16) bf16 qtile[16 * 512];
  int qt = blockIdx.x;             // 2048 query tiles (batches contiguous)
  int b = qt >> 8;
  int lane = threadIdx.x;
  for (int i = lane; i < 16 * 512 / 8; i += 32)
    stage16((const uint4*)(qb + (size_t)qt * 16 * Cq) + i, (uint4*)qtile + i);
  wait_async();
  __syncthreads();
  const bf16* kbase = kb + (size_t)b * Nq * Cq;
  float rmax[8], rsum[8];
#pragma unroll
  for (int r = 0; r < 8; ++r) { rmax[r] = -1e30f; rsum[r] = 0.f; }
  for (int nt = 0; nt < 256; ++nt) {
    v8f d = {};
    for (int kt = 0; kt < 16; ++kt) {
      v16bf a   = frag_rowmajor(qtile + kt * 32, 512, lane);
      v16bf bk_ = frag_rowmajor(kbase + (size_t)(nt * 16) * Cq + kt * 32, 512, lane);
      d = wmma_bf16(a, bk_, d);
    }
#pragma unroll
    for (int r = 0; r < 8; ++r) {
      float v = d[r] * SCALEF;
      float t = v;
      t = fmaxf(t, __shfl_xor(t, 1));
      t = fmaxf(t, __shfl_xor(t, 2));
      t = fmaxf(t, __shfl_xor(t, 4));
      t = fmaxf(t, __shfl_xor(t, 8));
      float mnew = fmaxf(rmax[r], t);
      float e = __expf(v - mnew);
      e += __shfl_xor(e, 1);
      e += __shfl_xor(e, 2);
      e += __shfl_xor(e, 4);
      e += __shfl_xor(e, 8);
      rsum[r] = rsum[r] * __expf(rmax[r] - mnew) + e;
      rmax[r] = mnew;
    }
  }
  if ((lane & 15) == 0) {
    int half = lane >> 4;
#pragma unroll
    for (int r = 0; r < 8; ++r) {
      int row = qt * 16 + half * 8 + r;
      ms[row * 2]     = rmax[r];
      ms[row * 2 + 1] = 1.0f / rsum[r];
    }
  }
}

// ---------------------------------------------------------------------------
// 6) Attention output (pass B): 4 waves per 16-query tile. Cooperative
//    scheme: wave w computes the score tile for ITS OWN 32-key slice and
//    publishes P (bf16) to LDS; after a barrier every wave consumes all 4
//    P tiles for its private 128-channel output chunk. This removes the
//    4x redundant score recomputation (scores dominate: 32 WMMA vs 8 for PV).
// ---------------------------------------------------------------------------
__global__ __launch_bounds__(128) void k_attn_out(const bf16* __restrict__ qb,
                                                  const bf16* __restrict__ kb,
                                                  const bf16* __restrict__ vtb,
                                                  const float* __restrict__ ms,
                                                  bf16* __restrict__ ao) {
  __shared__ __align__(16) bf16 qtile[16 * 512];
  __shared__ __align__(16) bf16 pbuf[4][16 * 32];
  int qt = blockIdx.x;
  int b = qt >> 8;
  int wave = threadIdx.x >> 5, lane = threadIdx.x & 31;
  for (int i = threadIdx.x; i < 16 * 512 / 8; i += 128)
    stage16((const uint4*)(qb + (size_t)qt * 16 * Cq) + i, (uint4*)qtile + i);
  wait_async();
  __syncthreads();
  int half = lane >> 4;
  float mrow[8], rsrow[8];
#pragma unroll
  for (int r = 0; r < 8; ++r) {
    int row = qt * 16 + half * 8 + r;
    mrow[r]  = ms[row * 2];
    rsrow[r] = ms[row * 2 + 1];
  }
  const bf16* kbase = kb + (size_t)b * Nq * Cq;
  const bf16* vbase = vtb + (size_t)b * Cq * Nq;
  v8f acc[8] = {};
  bf16* pb = &pbuf[wave][0];
  for (int kblock = 0; kblock < 32; ++kblock) {     // 128 keys per iteration
    int key0 = kblock * 128 + wave * 32;            // this wave's 32-key slice
    v8f d0 = {}, d1 = {};
    for (int kt = 0; kt < 16; ++kt) {
      v16bf a  = frag_rowmajor(qtile + kt * 32, 512, lane);
      v16bf b0 = frag_rowmajor(kbase + (size_t)key0 * Cq + kt * 32, 512, lane);
      v16bf b1 = frag_rowmajor(kbase + (size_t)(key0 + 16) * Cq + kt * 32, 512, lane);
      d0 = wmma_bf16(a, b0, d0);
      d1 = wmma_bf16(a, b1, d1);
    }
#pragma unroll
    for (int r = 0; r < 8; ++r) {
      float p0 = __expf(d0[r] * SCALEF - mrow[r]) * rsrow[r];
      float p1 = __expf(d1[r] * SCALEF - mrow[r]) * rsrow[r];
      int row = r + half * 8;
      pb[row * 32 + (lane & 15)]      = (bf16)p0;
      pb[row * 32 + 16 + (lane & 15)] = (bf16)p1;
    }
    __syncthreads();                                // publish P tiles
#pragma unroll
    for (int pt = 0; pt < 4; ++pt) {                // consume all 4 P tiles
      v16bf pa = frag_rowmajor(&pbuf[pt][0], 32, lane);
      int keyp = kblock * 128 + pt * 32;
#pragma unroll
      for (int ct = 0; ct < 8; ++ct) {
        v16bf vfr = frag_rowmajor(
            vbase + (size_t)(wave * 128 + ct * 16) * Nq + keyp, Nq, lane);
        acc[ct] = wmma_bf16(pa, vfr, acc[ct]);
      }
    }
    __syncthreads();                                // protect pbuf reuse (WAR)
  }
#pragma unroll
  for (int ct = 0; ct < 8; ++ct) {
    int ch = wave * 128 + ct * 16 + (lane & 15);
#pragma unroll
    for (int r = 0; r < 8; ++r) {
      int row = qt * 16 + r + half * 8;
      ao[(size_t)row * Cq + ch] = (bf16)acc[ct][r];
    }
  }
}

// ---------------------------------------------------------------------------
// 7) Output projection + bias + residual, fp32 store.
// ---------------------------------------------------------------------------
__global__ __launch_bounds__(256) void k_out_proj(const bf16* __restrict__ ao,
                                                  const bf16* __restrict__ wo,
                                                  const float* __restrict__ bo,
                                                  const float* __restrict__ x,
                                                  float* __restrict__ out) {
  __shared__ __align__(16) bf16 atile[16 * 512];
  int wave = threadIdx.x >> 5, lane = threadIdx.x & 31;
  int mt = blockIdx.x >> 2;
  int ct = ((blockIdx.x & 3) << 3) + wave;
  int m0 = mt << 4;
  for (int i = threadIdx.x; i < 16 * 512 / 8; i += 256)
    stage16((const uint4*)(ao + (size_t)m0 * Cq) + i, (uint4*)atile + i);
  wait_async();
  __syncthreads();
  v8f acc = {};
  for (int kt = 0; kt < 16; ++kt) {
    v16bf a = frag_rowmajor(atile + kt * 32, 512, lane);
    const v16bf* bp = (const v16bf*)(wo + (size_t)((kt * 32 + ct) * 32 + lane) * 16);
    acc = wmma_bf16(a, *bp, acc);
  }
  int half = lane >> 4;
  int col = (ct << 4) + (lane & 15);
  float bias = bo[col];
#pragma unroll
  for (int r = 0; r < 8; ++r) {
    int row = m0 + r + half * 8;
    size_t idx = (size_t)row * Cq + col;
    out[idx] = acc[r] + bias + x[idx];
  }
}

// ---------------------------------------------------------------------------
// Launch
// ---------------------------------------------------------------------------
extern "C" void kernel_launch(void* const* d_in, const int* in_sizes, int n_in,
                              void* d_out, int out_size, void* d_ws, size_t ws_size,
                              hipStream_t stream) {
  const float* x      = (const float*)d_in[0];
  const float* gscale = (const float*)d_in[1];
  const float* gbias  = (const float*)d_in[2];
  const float* wq     = (const float*)d_in[3];
  const float* bq     = (const float*)d_in[4];
  const float* wk     = (const float*)d_in[5];
  const float* bk     = (const float*)d_in[6];
  const float* wv     = (const float*)d_in[7];
  const float* bv     = (const float*)d_in[8];
  const float* wo     = (const float*)d_in[9];
  const float* bo     = (const float*)d_in[10];
  float* out = (float*)d_out;

  char* ws = (char*)d_ws;
  float* stats = (float*)ws;                               // 2 KB
  bf16* wpkq = (bf16*)(ws + 4096);                         // 4 x 512KB packed weights
  bf16* wpkk = (bf16*)(ws + 4096 + 1 * 524288);
  bf16* wpkv = (bf16*)(ws + 4096 + 2 * 524288);
  bf16* wpko = (bf16*)(ws + 4096 + 3 * 524288);
  size_t o = 4096 + 4 * 524288;
  bf16* hn  = (bf16*)(ws + o); o += 33554432;              // normalized acts / reused as attn out
  bf16* qb  = (bf16*)(ws + o); o += 33554432;              // Q row-major
  bf16* kb  = (bf16*)(ws + o); o += 33554432;              // K row-major
  bf16* vtb = (bf16*)(ws + o); o += 33554432;              // V transposed (C x N)
  float* ms = (float*)(ws + o);                            // per-row (max, 1/sum)

  k_gn_stats<<<256, 256, 0, stream>>>(x, stats);
  k_gn_apply<<<16384, 256, 0, stream>>>(x, stats, gscale, gbias, hn);
  k_pack_w<<<64, 256, 0, stream>>>(wq, wpkq);
  k_pack_w<<<64, 256, 0, stream>>>(wk, wpkk);
  k_pack_w<<<64, 256, 0, stream>>>(wv, wpkv);
  k_pack_w<<<64, 256, 0, stream>>>(wo, wpko);
  k_qkv<<<8192, 256, 0, stream>>>(hn, wpkq, wpkk, wpkv, bq, bk, bv, qb, kb, vtb);
  k_attn_stats<<<2048, 32, 0, stream>>>(qb, kb, ms);
  k_attn_out<<<2048, 128, 0, stream>>>(qb, kb, vtb, ms, hn);  // attn out overwrites hn
  k_out_proj<<<8192, 256, 0, stream>>>(hn, wpko, bo, x, out);
}
